// Attention_5257039970960
// MI455X (gfx1250) — matrix operand order repaired and checked by reading, not compiled, not run on hardware
//
#include <hip/hip_runtime.h>

#define NCTX   4096
#define DMODEL 2048
#define DHEAD  128

typedef unsigned short u16;
typedef __attribute__((ext_vector_type(16))) __bf16         bf16x16;
typedef __attribute__((ext_vector_type(8)))  float          f32x8;
typedef __attribute__((ext_vector_type(8)))  unsigned short u16x8;
typedef __attribute__((ext_vector_type(4)))  int            i32x4;

#define GLOBAL_AS __attribute__((address_space(1)))
#define LDS_AS    __attribute__((address_space(3)))

#if __has_builtin(__builtin_amdgcn_global_load_async_to_lds_b128)
  #define HAVE_ASYNC_LDS 1
#else
  #define HAVE_ASYNC_LDS 0
#endif

union FragU { bf16x16 v; u16x8 h[2]; };

__device__ __forceinline__ u16 f32_to_bf16_bits(float f) {
    unsigned u = __float_as_uint(f);
    u += 0x7FFFu + ((u >> 16) & 1u);      // round-to-nearest-even
    return (u16)(u >> 16);
}
__device__ __forceinline__ float bf16_bits_to_f32(u16 h) {
    return __uint_as_float(((unsigned)h) << 16);
}

__device__ __forceinline__ void wait_async0() {
#if HAVE_ASYNC_LDS
  #if __has_builtin(__builtin_amdgcn_s_wait_asynccnt)
    __builtin_amdgcn_s_wait_asynccnt(0);
  #else
    asm volatile("s_wait_asynccnt 0x0" ::: "memory");
  #endif
#endif
}

#if HAVE_ASYNC_LDS
__device__ __forceinline__ void g2l_b128(const u16* g, u16* l) {
    __builtin_amdgcn_global_load_async_to_lds_b128(
        (GLOBAL_AS i32x4*)g, (LDS_AS i32x4*)l, 0, 0);
}
#endif

// ---------------------------------------------------------------------------
// NT GEMM: C[M,N] = A[M,K] * B[N,K]^T, bf16 in, fp32 accumulate.
// 256 threads = 8 waves; macro tile 128M x 128N; wave (wm,wn) owns 32M x 64N
// = 2x4 WMMA fragments. A/B K-panels (128x32 bf16) double-buffered in LDS,
// copied with gfx1250 async-to-LDS when available.
// Fragment layouts per CDNA5 ISA 7.12.2:
//   A 16x32: lane L -> row M=L&15; v0-3 = K[(L>>4)*8..+7], v4-7 = +16
//   B 32x16: lane L -> col N=L&15; v0-3 = K[(L>>4)*8..+7], v4-7 = +16 (same as A)
// ---------------------------------------------------------------------------
#define LDSTRIDE 40   // halves per 32-half panel row (80B: 16B-aligned, staggers banks)

template <bool CAUSAL, bool OUT_BF16>
__global__ __launch_bounds__(256) void gemm_nt_wmma_bf16(
        const u16* __restrict__ A, const u16* __restrict__ B,
        void* __restrict__ Cout, int M, int N, int K) {
    __shared__ u16 As[2][128 * LDSTRIDE];
    __shared__ u16 Bs[2][128 * LDSTRIDE];

    const int t    = threadIdx.x;
    const int lane = t & 31;
    const int wave = t >> 5;
    const int wm   = wave & 3;      // 32-row strip within macro tile
    const int wn   = wave >> 2;     // 64-col half within macro tile
    const int rowWG = blockIdx.x * 128;
    const int colWG = blockIdx.y * 128;

    if (CAUSAL && colWG > rowWG + 127) {       // block strictly above diagonal
        u16* C = (u16*)Cout;
        for (int i = t; i < 128 * 128; i += 256) {
            const int r = i >> 7, c = i & 127;
            C[(size_t)(rowWG + r) * N + colWG + c] = 0xFF80u;   // bf16 -inf
        }
        return;
    }

    // Per-thread copy assignment: each panel is 512 x 16B chunks (128 rows x 4);
    // thread handles chunks t and t+256.
    const int r0c = t >> 2,         c0c = (t & 3) * 8;
    const int r1c = (t + 256) >> 2, c1c = ((t + 256) & 3) * 8;
    const u16* Ag = A + (size_t)rowWG * K;
    const u16* Bg = B + (size_t)colWG * K;

    auto stage = [&](int buf, int k0) {
#if HAVE_ASYNC_LDS
        g2l_b128(Ag + (size_t)r0c * K + k0 + c0c, &As[buf][r0c * LDSTRIDE + c0c]);
        g2l_b128(Ag + (size_t)r1c * K + k0 + c1c, &As[buf][r1c * LDSTRIDE + c1c]);
        g2l_b128(Bg + (size_t)r0c * K + k0 + c0c, &Bs[buf][r0c * LDSTRIDE + c0c]);
        g2l_b128(Bg + (size_t)r1c * K + k0 + c1c, &Bs[buf][r1c * LDSTRIDE + c1c]);
#else
        *(u16x8*)&As[buf][r0c * LDSTRIDE + c0c] = *(const u16x8*)(Ag + (size_t)r0c * K + k0 + c0c);
        *(u16x8*)&As[buf][r1c * LDSTRIDE + c1c] = *(const u16x8*)(Ag + (size_t)r1c * K + k0 + c1c);
        *(u16x8*)&Bs[buf][r0c * LDSTRIDE + c0c] = *(const u16x8*)(Bg + (size_t)r0c * K + k0 + c0c);
        *(u16x8*)&Bs[buf][r1c * LDSTRIDE + c1c] = *(const u16x8*)(Bg + (size_t)r1c * K + k0 + c1c);
#endif
    };

    f32x8 acc[2][4] = {};

    stage(0, 0);
    wait_async0();
    __syncthreads();

    const int steps = K >> 5;               // K / 32
    for (int s = 0; s < steps; ++s) {
        const int cur = s & 1;
        if (s + 1 < steps) stage(cur ^ 1, (s + 1) << 5);

        #pragma unroll
        for (int mt = 0; mt < 2; ++mt) {
            FragU a;
            const u16* pa = &As[cur][(wm * 32 + mt * 16 + (lane & 15)) * LDSTRIDE
                                     + ((lane >> 4) << 3)];
            a.h[0] = *(const u16x8*)pa;
            a.h[1] = *(const u16x8*)(pa + 16);
            #pragma unroll
            for (int nt = 0; nt < 4; ++nt) {
                FragU b;
                const u16* pb = &Bs[cur][(wn * 64 + nt * 16 + (lane & 15)) * LDSTRIDE
                                         + ((lane >> 4) << 3)];
                b.h[0] = *(const u16x8*)pb;
                b.h[1] = *(const u16x8*)(pb + 16);
                acc[mt][nt] = __builtin_amdgcn_wmma_f32_16x16x32_bf16(
                                  false, a.v, false, b.v, (short)0, acc[mt][nt],
                                  false, false);
            }
        }
        wait_async0();
        __syncthreads();
    }

    const float NEG_INF = -__builtin_inff();
    const int nn = lane & 15;
    const int mb = (lane >> 4) << 3;
    #pragma unroll
    for (int mt = 0; mt < 2; ++mt)
        #pragma unroll
        for (int nt = 0; nt < 4; ++nt)
            #pragma unroll
            for (int r = 0; r < 8; ++r) {
                const int grow = rowWG + wm * 32 + mt * 16 + mb + r;
                const int gcol = colWG + wn * 64 + nt * 16 + nn;
                float v = acc[mt][nt][r];
                if (CAUSAL && gcol > grow) v = NEG_INF;
                if constexpr (OUT_BF16)
                    ((u16*)Cout)[(size_t)grow * N + gcol] = f32_to_bf16_bits(v);
                else
                    ((float*)Cout)[(size_t)grow * N + gcol] = v;
            }
}

// ---------------------------------------------------------------------------
// fp32 -> bf16 elementwise (vec4)
// ---------------------------------------------------------------------------
__global__ __launch_bounds__(256) void cvt_f32_to_bf16(
        const float* __restrict__ in, u16* __restrict__ out, int n4) {
    const int i = blockIdx.x * 256 + threadIdx.x;
    if (i >= n4) return;
    const float4 v = ((const float4*)in)[i];
    union { u16 u[4]; unsigned long long ll; } o;
    o.u[0] = f32_to_bf16_bits(v.x);
    o.u[1] = f32_to_bf16_bits(v.y);
    o.u[2] = f32_to_bf16_bits(v.z);
    o.u[3] = f32_to_bf16_bits(v.w);
    ((unsigned long long*)out)[i] = o.ll;
}

// ---------------------------------------------------------------------------
// LDS-tiled transpose: X[M,N] fp32 -> XT[N,M] bf16 (for V^T operand)
// ---------------------------------------------------------------------------
__global__ __launch_bounds__(256) void transpose_f32_to_bf16(
        const float* __restrict__ X, u16* __restrict__ XT, int M, int N) {
    __shared__ float tile[32][33];
    const int tx = threadIdx.x & 31;
    const int ty = threadIdx.x >> 5;
    const int r0 = blockIdx.x * 32;
    const int c0 = blockIdx.y * 32;
    #pragma unroll
    for (int i = ty; i < 32; i += 8)
        tile[i][tx] = X[(size_t)(r0 + i) * N + (c0 + tx)];
    __syncthreads();
    #pragma unroll
    for (int i = ty; i < 32; i += 8)
        XT[(size_t)(c0 + i) * M + (r0 + tx)] = f32_to_bf16_bits(tile[tx][i]);
}

// ---------------------------------------------------------------------------
// Row softmax, bf16 in place. One WG (8 waves) per row; row cached in LDS.
// Masked entries are bf16 -inf -> exp()=0.
// ---------------------------------------------------------------------------
__global__ __launch_bounds__(256) void softmax_row_bf16_inplace(
        u16* __restrict__ S, int N) {
    __shared__ float row[NCTX];
    __shared__ float red[8];
    const int t    = threadIdx.x;
    const int lane = t & 31;
    const int wv   = t >> 5;
    u16* r = S + (size_t)blockIdx.x * N;

    float lmax = -__builtin_inff();
    for (int c = t; c < N; c += 256) {
        const float v = bf16_bits_to_f32(r[c]);
        row[c] = v;
        lmax = fmaxf(lmax, v);
    }
    #pragma unroll
    for (int off = 16; off > 0; off >>= 1)
        lmax = fmaxf(lmax, __shfl_xor(lmax, off, 32));
    if (lane == 0) red[wv] = lmax;
    __syncthreads();
    float gmax = red[0];
    #pragma unroll
    for (int w = 1; w < 8; ++w) gmax = fmaxf(gmax, red[w]);
    __syncthreads();

    float lsum = 0.0f;
    for (int c = t; c < N; c += 256) {
        const float e = __expf(row[c] - gmax);
        row[c] = e;
        lsum += e;
    }
    #pragma unroll
    for (int off = 16; off > 0; off >>= 1)
        lsum += __shfl_xor(lsum, off, 32);
    if (lane == 0) red[wv] = lsum;
    __syncthreads();
    float gsum = 0.0f;
    #pragma unroll
    for (int w = 0; w < 8; ++w) gsum += red[w];
    const float inv = 1.0f / gsum;
    for (int c = t; c < N; c += 256)
        r[c] = f32_to_bf16_bits(row[c] * inv);
}

// ---------------------------------------------------------------------------
extern "C" void kernel_launch(void* const* d_in, const int* in_sizes, int n_in,
                              void* d_out, int out_size, void* d_ws, size_t ws_size,
                              hipStream_t stream) {
    (void)in_sizes; (void)n_in; (void)out_size; (void)ws_size;
    const float* x  = (const float*)d_in[0];
    const float* Wk = (const float*)d_in[1];
    const float* Wq = (const float*)d_in[2];
    const float* W2 = (const float*)d_in[3];
    float* out = (float*)d_out;

    char* ws = (char*)d_ws;
    size_t off = 0;
    auto carve = [&](size_t bytes) -> void* {
        void* p = ws + off;
        off = (off + bytes + 255) & ~(size_t)255;
        return p;
    };
    u16* xb  = (u16*)carve((size_t)NCTX * DMODEL * 2);   // x bf16 row-major
    u16* xbT = (u16*)carve((size_t)DMODEL * NCTX * 2);   // x^T bf16 (V^T)
    u16* wqb = (u16*)carve((size_t)DHEAD * DMODEL * 2);
    u16* wkb = (u16*)carve((size_t)DHEAD * DMODEL * 2);
    u16* w2b = (u16*)carve((size_t)DMODEL * DMODEL * 2);
    u16* qb  = (u16*)carve((size_t)NCTX * DHEAD * 2);
    u16* kb  = (u16*)carve((size_t)NCTX * DHEAD * 2);
    u16* Sb  = (u16*)carve((size_t)NCTX * NCTX * 2);     // scores -> probs (32 MB, L2-resident)
    u16* ob  = (u16*)carve((size_t)NCTX * DMODEL * 2);   // attn @ x, bf16

    // ---- downcasts ----
    cvt_f32_to_bf16<<<(NCTX * DMODEL / 4 + 255) / 256, 256, 0, stream>>>(x,  xb,  NCTX * DMODEL / 4);
    cvt_f32_to_bf16<<<(DHEAD * DMODEL / 4 + 255) / 256, 256, 0, stream>>>(Wq, wqb, DHEAD * DMODEL / 4);
    cvt_f32_to_bf16<<<(DHEAD * DMODEL / 4 + 255) / 256, 256, 0, stream>>>(Wk, wkb, DHEAD * DMODEL / 4);
    cvt_f32_to_bf16<<<(DMODEL * DMODEL / 4 + 255) / 256, 256, 0, stream>>>(W2, w2b, DMODEL * DMODEL / 4);
    transpose_f32_to_bf16<<<dim3(NCTX / 32, DMODEL / 32), 256, 0, stream>>>(x, xbT, NCTX, DMODEL);

    // ---- q = x @ Wq^T, k = x @ Wk^T  (M=4096, N=128, K=2048) ----
    gemm_nt_wmma_bf16<false, true><<<dim3(NCTX / 128, DHEAD / 128), 256, 0, stream>>>(xb, wqb, qb, NCTX, DHEAD, DMODEL);
    gemm_nt_wmma_bf16<false, true><<<dim3(NCTX / 128, DHEAD / 128), 256, 0, stream>>>(xb, wkb, kb, NCTX, DHEAD, DMODEL);

    // ---- S = causal(q @ k^T)  (M=N=4096, K=128) ----
    gemm_nt_wmma_bf16<true, true><<<dim3(NCTX / 128, NCTX / 128), 256, 0, stream>>>(qb, kb, Sb, NCTX, NCTX, DHEAD);

    // ---- P = softmax_rows(S), in place ----
    softmax_row_bf16_inplace<<<NCTX, 256, 0, stream>>>(Sb, NCTX);

    // ---- O = P @ x  (B = x^T, M=4096, N=2048, K=4096) ----
    gemm_nt_wmma_bf16<false, true><<<dim3(NCTX / 128, DMODEL / 128), 256, 0, stream>>>(Sb, xbT, ob, NCTX, DMODEL, NCTX);

    // ---- out = O @ W2^T  (fp32 output, M=4096, N=2048, K=2048) ----
    gemm_nt_wmma_bf16<false, false><<<dim3(NCTX / 128, DMODEL / 128), 256, 0, stream>>>(ob, w2b, out, NCTX, DMODEL, DMODEL);
}
